// AttentionAgents_29746943492400
// MI455X (gfx1250) — matrix operand order repaired and checked by reading, not compiled, not run on hardware
//
#include <hip/hip_runtime.h>
#include <hip/hip_bf16.h>
#include <math.h>

typedef __attribute__((ext_vector_type(16))) __bf16 v16bf;
typedef __attribute__((ext_vector_type(8)))  __bf16 v8bf;
typedef __attribute__((ext_vector_type(8)))  float  v8f;

#define NPOS 4096   // 64*64 spatial positions
#define NB   128    // batch

union V16 { v16bf v; struct { v8bf lo, hi; } h; };

// single packed f32->bf16 convert (mnemonic verified in this toolchain's output)
__device__ __forceinline__ unsigned pack2bf(float a, float b) {
  unsigned r;
  asm("v_cvt_pk_bf16_f32 %0, %1, %2" : "=v"(r) : "v"(a), "v"(b));
  return r;
}

// ---------------------------------------------------------------------------
// Fixed cos-product Fourier spatial basis: S[p][u*8+v], p = h*64+w
// ---------------------------------------------------------------------------
__global__ void spatial_basis_kernel(float* __restrict__ S) {
  int p = blockIdx.x * blockDim.x + threadIdx.x;
  if (p >= NPOS) return;
  int i = p >> 6, j = p & 63;
  const float PI = 3.14159265358979323846f;
  float ph = (float)(i + 1) * (PI / 64.0f);
  float pw = (float)(j + 1) * (PI / 64.0f);
  float ca[8], cb[8];
#pragma unroll
  for (int u = 0; u < 8; ++u) { ca[u] = cosf(ph * (u + 1)); cb[u] = cosf(pw * (u + 1)); }
#pragma unroll
  for (int u = 0; u < 8; ++u)
#pragma unroll
    for (int v = 0; v < 8; ++v)
      S[(size_t)p * 64 + u * 8 + v] = ca[u] * cb[v];
}

// ---------------------------------------------------------------------------
// One-time weight pack: f32 (K,N) [or (N,K) if transW] -> bf16 in WMMA
// B-operand lane layout, zero-padded. out[((nt*Kt + kt)*32 + lane)*16 + i]
// ---------------------------------------------------------------------------
__global__ void pack_weight_kernel(const float* __restrict__ W, int ldw, int transW,
                                   const int* __restrict__ colBasePtr, int colScale,
                                   int K, int N, int Kt, int Nt,
                                   __bf16* __restrict__ out) {
  int idx = blockIdx.x * blockDim.x + threadIdx.x;
  int total = Nt * Kt * 512;
  if (idx >= total) return;
  int i    = idx & 15;
  int lane = (idx >> 4) & 31;
  int kt   = (idx >> 9) % Kt;
  int nt   = (idx >> 9) / Kt;
  int n = nt * 16 + (lane & 15);
  int k = kt * 32 + ((lane >> 4) << 4) + i;
  int colBase = colBasePtr ? colBasePtr[0] * colScale : 0;
  float v = 0.0f;
  if (k < K && n < N)
    v = transW ? W[(size_t)(colBase + n) * ldw + k]
               : W[(size_t)k * ldw + colBase + n];
  out[idx] = (__bf16)v;
}

// ---------------------------------------------------------------------------
// Pack per-batch Q chunk (4x72 f32) into bf16 B-operand layout (96K x 16N pad)
// ---------------------------------------------------------------------------
__global__ void pack_q_kernel(const float* __restrict__ Qc, __bf16* __restrict__ out) {
  int idx = blockIdx.x * blockDim.x + threadIdx.x;
  if (idx >= NB * 3 * 512) return;
  int i    = idx & 15;
  int lane = (idx >> 4) & 31;
  int s    = (idx >> 9) % 3;
  int b    = (idx >> 9) / 3;
  int q = lane & 15;
  int k = s * 32 + ((lane >> 4) << 4) + i;
  float v = (q < 4 && k < 72) ? Qc[(size_t)b * 288 + q * 72 + k] : 0.0f;
  out[idx] = (__bf16)v;
}

// ---------------------------------------------------------------------------
// C = act(A @ Bpacked + bias[colBase+n]) [+= C]; bf16 WMMA, f32 accumulate.
// Double-buffered bf16 A tile in LDS: one barrier per k-step; staging of the
// next tile overlaps the current WMMA. No bounds checks in K loop.
// ---------------------------------------------------------------------------
__global__ void gemm_wmma(const float* __restrict__ A, int lda,
                          const __bf16* __restrict__ Bp, int Kt,
                          const float* __restrict__ bias,
                          float* __restrict__ C, int ldc,
                          int M, int N, int act, int accum,
                          const int* __restrict__ colBasePtr, int colScale) {
  __shared__ unsigned ldsA32[2][16 * 16];      // 2 x (16 rows x 32 bf16)
  const int tid  = threadIdx.x;
  const int lane = tid & 31;
  const int wave = tid >> 5;
  const int m0 = blockIdx.y * 16;
  const int n0 = blockIdx.x * 64 + wave * 16;
  const int colBase = colBasePtr ? colBasePtr[0] * colScale : 0;

  const int arow  = lane & 15;
  const int aKoff = (lane >> 4) << 4;          // 0 / 16 (contiguous k layout)
  const __bf16* bp = Bp + (size_t)(n0 >> 4) * Kt * 512 + lane * 16;

  auto stageA = [&](int kt, int buf) {
#pragma unroll
    for (int u = 0; u < 2; ++u) {
      int j = tid + u * 128;
      int r = j >> 4, cp = j & 15;
      float2 f = *(const float2*)&A[(size_t)(m0 + r) * lda + kt * 32 + cp * 2];
      ldsA32[buf][r * 16 + cp] = pack2bf(f.x, f.y);
    }
  };

  v8f acc = {};
  stageA(0, 0);
  __syncthreads();
  for (int kt = 0; kt < Kt; ++kt) {
    int cur = kt & 1;
    if (kt + 1 < Kt) stageA(kt + 1, cur ^ 1);
    const __bf16* ab = (const __bf16*)ldsA32[cur];
    V16 ua;
    ua.h.lo = *(const v8bf*)&ab[arow * 32 + aKoff];
    ua.h.hi = *(const v8bf*)&ab[arow * 32 + aKoff + 8];
    v16bf bv = *(const v16bf*)(bp + (size_t)kt * 512);
    acc = __builtin_amdgcn_wmma_f32_16x16x32_bf16(false, ua.v, false, bv,
                                                  (short)0, acc, false, false);
    __syncthreads();
  }

  const int mh = (lane >> 4) << 3;
  const int n  = n0 + (lane & 15);
#pragma unroll
  for (int i = 0; i < 8; ++i) {
    int m = m0 + mh + i;
    if (m < M && n < N) {
      float v = acc[i] + bias[colBase + n];
      if (act == 1) v = fmaxf(v, 0.0f);
      if (accum) v += C[(size_t)m * ldc + n];
      C[(size_t)m * ldc + n] = v;
    }
  }
}

// ---------------------------------------------------------------------------
// Attention: one block (256 thr = 8 waves) per batch element.
// Pass 1: scores = K . Q^T via wave-private bf16 LDS K-tiles.
// Softmax over 4096 (probabilities stored bf16). Pass 2: a = P . V with
// double-buffered 64-row transposed bf16 V-tiles; one barrier per stage.
// ---------------------------------------------------------------------------
__global__ void attn_wmma(const float* __restrict__ x,    // (128, 4096, 128)
                          const float* __restrict__ S,    // (4096, 64)
                          const __bf16* __restrict__ qpk, // packed Q operands
                          float* __restrict__ aOut)       // (128, 4, 184)
{
  extern __shared__ char smemraw[];
  float*    probs = (float*)smemraw;                           // 65536 B
  __bf16*   pbf   = (__bf16*)(smemraw + 65536);                // 32768 B
  unsigned* tile0 = (unsigned*)(smemraw + 65536 + 32768);      // 25600 B
  unsigned* tile1 = tile0 + 6400;                              // 25600 B
  float*    red   = (float*)(smemraw + 65536 + 32768 + 51200); // 1024 B

  const int batch = blockIdx.x;
  const int tid  = threadIdx.x;
  const int lane = tid & 31;
  const int wave = tid >> 5;
  const float* xb = x + (size_t)batch * NPOS * 128;

  const int arow  = lane & 15;
  const int aKoff = (lane >> 4) << 4;
  v16bf zero16 = {};

  // ---- Pass 1: wave-private bf16 K-tile (16 x 96), no block barriers ----
  {
    unsigned* kt32 = tile0 + wave * 768;       // 16 rows x 48 u32
    const __bf16* kt16 = (const __bf16*)kt32;
    for (int idx = lane; idx < 16 * 12; idx += 32)   // zero pad cols 72..95
      kt32[(idx / 12) * 48 + 36 + (idx % 12)] = 0u;

    v16bf bq[3];
#pragma unroll
    for (int s = 0; s < 3; ++s)
      bq[s] = *(const v16bf*)(qpk + (((size_t)batch * 3 + s) * 32 + lane) * 16);

    for (int t = wave; t < NPOS / 16; t += 8) {
      int p0 = t * 16;
      for (int idx = lane; idx < 64; idx += 32) {    // keys 16x8 -> u32 0..3
        int r = idx >> 2, cp = idx & 3;
        float2 f = *(const float2*)&xb[(size_t)(p0 + r) * 128 + cp * 2];
        kt32[r * 48 + cp] = pack2bf(f.x, f.y);
      }
      for (int idx = lane; idx < 512; idx += 32) {   // basis 16x64 -> u32 4..35
        int r = idx >> 5, cp = idx & 31;
        float2 f = *(const float2*)&S[(size_t)(p0 + r) * 64 + cp * 2];
        kt32[r * 48 + 4 + cp] = pack2bf(f.x, f.y);
      }
      v8f acc = {};
#pragma unroll
      for (int s = 0; s < 3; ++s) {
        V16 ua;
        ua.h.lo = *(const v8bf*)&kt16[arow * 96 + s * 32 + aKoff];
        ua.h.hi = *(const v8bf*)&kt16[arow * 96 + s * 32 + aKoff + 8];
        acc = __builtin_amdgcn_wmma_f32_16x16x32_bf16(false, ua.v, false, bq[s],
                                                      (short)0, acc, false, false);
      }
      int q = lane & 15;
      if (q < 4) {
        int mh = (lane >> 4) << 3;
#pragma unroll
        for (int i = 0; i < 8; ++i)
          probs[q * NPOS + p0 + mh + i] = acc[i];
      }
    }
  }
  __syncthreads();

  // ---------------- softmax; result stored as bf16 in pbf ----------------
  for (int q = 0; q < 4; ++q) {
    float m = -3.4e38f;
    for (int i = tid; i < NPOS; i += 256) m = fmaxf(m, probs[q * NPOS + i]);
    red[tid] = m; __syncthreads();
    for (int s = 128; s > 0; s >>= 1) {
      if (tid < s) red[tid] = fmaxf(red[tid], red[tid + s]);
      __syncthreads();
    }
    float gm = red[0]; __syncthreads();
    float sum = 0.0f;
    for (int i = tid; i < NPOS; i += 256) {
      float e = expf(probs[q * NPOS + i] - gm);
      probs[q * NPOS + i] = e; sum += e;
    }
    red[tid] = sum; __syncthreads();
    for (int s = 128; s > 0; s >>= 1) {
      if (tid < s) red[tid] += red[tid + s];
      __syncthreads();
    }
    float inv = 1.0f / red[0]; __syncthreads();
    for (int i = tid; i < NPOS; i += 256)
      pbf[q * NPOS + i] = (__bf16)(probs[q * NPOS + i] * inv);
    __syncthreads();
  }

  // ---- Pass 2: double-buffered 64-row V-tiles, transposed bf16 [200ch][64r]
  auto stageV = [&](int kb, unsigned* vt) {
    for (int idx = tid; idx < 4096; idx += 256) {        // x: 64 rows x 128 ch
      int c = idx & 127, rp = idx >> 7;
      const float* s0 = &xb[(size_t)(kb + rp * 2) * 128 + c];
      vt[c * 32 + rp] = pack2bf(s0[0], s0[128]);
    }
    for (int idx = tid; idx < 2048; idx += 256) {        // S: 64 rows x 64 ch
      int c = idx & 63, rp = idx >> 6;
      const float* s0 = &S[(size_t)(kb + rp * 2) * 64 + c];
      vt[(128 + c) * 32 + rp] = pack2bf(s0[0], s0[64]);
    }
  };
  for (int idx = tid; idx < 8 * 32; idx += 256) {        // zero pad ch 192..199
    tile0[(192 + (idx >> 5)) * 32 + (idx & 31)] = 0u;
    tile1[(192 + (idx >> 5)) * 32 + (idx & 31)] = 0u;
  }
  stageV(0, tile0);
  __syncthreads();

  v8f acc0 = {}, acc1 = {};
  const int nt0  = wave;
  const int nt1  = wave + 8;
  const int bcol = lane & 15;
  const int bKoff = (lane >> 4) << 4;
  const int n8a = 8 + nt0 * 16 + bcol;         // unified V channel index
  const int n8b = 8 + nt1 * 16 + bcol;
  const int prow = (arow & 3) * NPOS;

  for (int kb = 0; kb < NPOS; kb += 64) {
    unsigned* cur = ((kb >> 6) & 1) ? tile1 : tile0;
    unsigned* nxt = ((kb >> 6) & 1) ? tile0 : tile1;
    if (kb + 64 < NPOS) {
      stageV(kb + 64, nxt);
      __builtin_prefetch(&xb[(size_t)(kb + 128) * 128 + tid * 32], 0, 1);
    }
    const __bf16* vt16 = (const __bf16*)cur;
#pragma unroll
    for (int ks = 0; ks < 64; ks += 32) {
      V16 ua;
      ua.h.lo = *(const v8bf*)&pbf[prow + kb + ks + aKoff];
      ua.h.hi = *(const v8bf*)&pbf[prow + kb + ks + aKoff + 8];
      if (arow >= 4) ua.v = zero16;
      v16bf bv0 = *(const v16bf*)&vt16[n8a * 64 + ks + bKoff];
      acc0 = __builtin_amdgcn_wmma_f32_16x16x32_bf16(false, ua.v, false, bv0,
                                                     (short)0, acc0, false, false);
      if (nt1 < 12) {
        v16bf bv1 = *(const v16bf*)&vt16[n8b * 64 + ks + bKoff];
        acc1 = __builtin_amdgcn_wmma_f32_16x16x32_bf16(false, ua.v, false, bv1,
                                                       (short)0, acc1, false, false);
      }
    }
    __syncthreads();
  }

  if (lane < 16) {
#pragma unroll
    for (int i = 0; i < 4; ++i) {
      int d0 = nt0 * 16 + bcol;
      if (d0 < 184) aOut[(size_t)batch * 736 + i * 184 + d0] = acc0[i];
      if (nt1 < 12) {
        int d1 = nt1 * 16 + bcol;
        if (d1 < 184) aOut[(size_t)batch * 736 + i * 184 + d1] = acc1[i];
      }
    }
  }
}

// ---------------------------------------------------------------------------
// answer (128 x 1056, zero padded) = [a(736) | Q(288) | r_prev | a_prev | 0..]
// ---------------------------------------------------------------------------
__global__ void assemble_answer(const float* __restrict__ aAns,
                                const float* __restrict__ Qc,
                                const float* __restrict__ r_prev,
                                const float* __restrict__ a_prev,
                                float* __restrict__ answer) {
  int idx = blockIdx.x * blockDim.x + threadIdx.x;
  if (idx >= NB * 1056) return;
  int n = idx / 1056, c = idx % 1056;
  float v;
  if (c < 736)        v = aAns[(size_t)n * 736 + c];
  else if (c < 1024)  v = Qc[(size_t)n * 288 + (c - 736)];
  else if (c == 1024) v = r_prev[n];
  else if (c == 1025) v = a_prev[n];
  else                v = 0.0f;
  answer[idx] = v;
}

// ---------------------------------------------------------------------------
__global__ void lstm_elem(const float* __restrict__ gates,
                          const float* __restrict__ prev_cell,
                          float* __restrict__ hN) {
  int idx = blockIdx.x * blockDim.x + threadIdx.x;
  if (idx >= NB * 256) return;
  int n = idx >> 8, j = idx & 255;
  const float* g = gates + (size_t)n * 1024;
  float ig = 1.0f / (1.0f + expf(-g[j]));
  float fg = 1.0f / (1.0f + expf(-g[256 + j]));
  float gg = tanhf(g[512 + j]);
  float og = 1.0f / (1.0f + expf(-g[768 + j]));
  float c2 = fg * prev_cell[idx] + ig * gg;
  hN[idx] = og * tanhf(c2);
}

// ---------------------------------------------------------------------------
extern "C" void kernel_launch(void* const* d_in, const int* in_sizes, int n_in,
                              void* d_out, int out_size, void* d_ws, size_t ws_size,
                              hipStream_t stream) {
  (void)in_sizes; (void)n_in; (void)out_size; (void)ws_size;
  const float* x      = (const float*)d_in[0];
  const int*   cptr   = (const int*)d_in[1];
  const float* r_prev = (const float*)d_in[2];
  const float* a_prev = (const float*)d_in[3];
  const float* ph     = (const float*)d_in[4];
  const float* pc     = (const float*)d_in[5];
  const float* q_w0 = (const float*)d_in[6];  const float* q_b0 = (const float*)d_in[7];
  const float* q_w1 = (const float*)d_in[8];  const float* q_b1 = (const float*)d_in[9];
  const float* q_w2 = (const float*)d_in[10]; const float* q_b2 = (const float*)d_in[11];
  const float* a_w0 = (const float*)d_in[12]; const float* a_b0 = (const float*)d_in[13];
  const float* a_w1 = (const float*)d_in[14]; const float* a_b1 = (const float*)d_in[15];
  const float* a_w2 = (const float*)d_in[16]; const float* a_b2 = (const float*)d_in[17];
  const float* w_ih = (const float*)d_in[18]; const float* w_hh = (const float*)d_in[19];
  const float* b_ih = (const float*)d_in[20]; const float* b_hh = (const float*)d_in[21];
  const float* p_w  = (const float*)d_in[22]; const float* p_b  = (const float*)d_in[23];
  const float* v_w  = (const float*)d_in[24]; const float* v_b  = (const float*)d_in[25];
  float* out = (float*)d_out;

  // ---------------- workspace layout ----------------
  float* ws     = (float*)d_ws;
  float* S      = ws;                  // 262144
  float* H1     = S + 262144;          // 65536 (reused for H3)
  float* H2     = H1 + 65536;          // 65536 (reused for H4)
  float* Qc     = H2 + 65536;          // 36864
  float* aAns   = Qc + 36864;          // 94208
  float* answer = aAns + 94208;        // 135168 (128 x 1056, padded)
  float* gates  = answer + 135168;     // 131072
  float* ans256 = gates + 131072;      // 32768
  float* hN     = ans256 + 32768;      // 32768
  __bf16* pk = (__bf16*)(hN + 32768);  // bf16 packed weights
  __bf16* pq0 = pk;                    // Kt=8,  Nt=32 : 131072
  __bf16* pq1 = pq0 + 131072;          // Kt=16, Nt=32 : 262144
  __bf16* pq2 = pq1 + 262144;          // Kt=16, Nt=20 : 163840
  __bf16* pa0 = pq2 + 163840;          // Kt=33, Nt=32 : 540672
  __bf16* pa1 = pa0 + 540672;          // Kt=16, Nt=32 : 262144
  __bf16* pa2 = pa1 + 262144;          // Kt=16, Nt=16 : 131072
  __bf16* pih = pa2 + 131072;          // Kt=8,  Nt=64 : 262144
  __bf16* phh = pih + 262144;          // Kt=8,  Nt=64 : 262144
  __bf16* ppw = phh + 262144;          // Kt=8,  Nt=4  : 16384
  __bf16* pvw = ppw + 16384;           // Kt=8,  Nt=4  : 16384
  __bf16* pqc = pvw + 16384;           // 128*3*512    : 196608

  auto pack = [&](const float* W, int ldw, int transW, const int* cb, int cs,
                  int K, int N, int Kt, int Nt, __bf16* o) {
    int total = Nt * Kt * 512;
    pack_weight_kernel<<<dim3((total + 255) / 256), dim3(256), 0, stream>>>(
        W, ldw, transW, cb, cs, K, N, Kt, Nt, o);
  };
  auto gemm = [&](const float* A, int lda, const __bf16* Bp, int Kt,
                  const float* bias, float* C, int ldc, int M, int N,
                  int act, int accum, const int* cb, int cs) {
    dim3 grid((N + 63) / 64, (M + 15) / 16);
    gemm_wmma<<<grid, dim3(128), 0, stream>>>(A, lda, Bp, Kt, bias, C, ldc,
                                              M, N, act, accum, cb, cs);
  };

  // one-time packs + basis (cheap; ordered on stream)
  spatial_basis_kernel<<<dim3(16), dim3(256), 0, stream>>>(S);
  pack(q_w0, 512, 0, nullptr, 0,  256,  512,  8, 32, pq0);
  pack(q_w1, 512, 0, nullptr, 0,  512,  512, 16, 32, pq1);
  pack(q_w2, 1152, 0, cptr, 288,  512,  288, 16, 20, pq2);
  pack(a_w0, 512, 0, nullptr, 0, 1026,  512, 33, 32, pa0);
  pack(a_w1, 512, 0, nullptr, 0,  512,  512, 16, 32, pa1);
  pack(a_w2, 256, 0, nullptr, 0,  512,  256, 16, 16, pa2);
  pack(w_ih, 256, 1, nullptr, 0,  256, 1024,  8, 64, pih);
  pack(w_hh, 256, 1, nullptr, 0,  256, 1024,  8, 64, phh);
  pack(p_w,   18, 0, nullptr, 0,  256,   18,  8,  4, ppw);
  pack(v_w,   18, 0, nullptr, 0,  256,   18,  8,  4, pvw);

  // Q-MLP: 256 -> 512 -> 512 -> slice c*288..+288
  gemm(ph, 256, pq0,  8, q_b0, H1, 512, NB, 512, 1, 0, nullptr, 0);
  gemm(H1, 512, pq1, 16, q_b1, H2, 512, NB, 512, 1, 0, nullptr, 0);
  gemm(H2, 512, pq2, 16, q_b2, Qc, 288, NB, 288, 0, 0, cptr, 288);
  pack_q_kernel<<<dim3((NB * 3 * 512 + 255) / 256), dim3(256), 0, stream>>>(Qc, pqc);

  // Spatial attention: streams the 256MB x tensor once through LDS
  size_t smem = 65536 + 32768 + 51200 + 1024;   // 150528 B dynamic LDS
  attn_wmma<<<dim3(NB), dim3(256), smem, stream>>>(x, S, pqc, aAns);

  assemble_answer<<<dim3((NB * 1056 + 255) / 256), dim3(256), 0, stream>>>(
      aAns, Qc, r_prev, a_prev, answer);

  // answer-MLP: 1026(pad 1056) -> 512 -> 512 -> 256
  gemm(answer, 1056, pa0, 33, a_b0, H1, 512, NB, 512, 1, 0, nullptr, 0);
  gemm(H1, 512, pa1, 16, a_b1, H2, 512, NB, 512, 1, 0, nullptr, 0);
  gemm(H2, 512, pa2, 16, a_b2, ans256, 256, NB, 256, 0, 0, nullptr, 0);

  // LSTM gates = x@W_ih^T + b_ih + h@W_hh^T + b_hh
  gemm(ans256, 256, pih, 8, b_ih, gates, 1024, NB, 1024, 0, 0, nullptr, 0);
  gemm(ph,     256, phh, 8, b_hh, gates, 1024, NB, 1024, 0, 1, nullptr, 0);
  lstm_elem<<<dim3(NB), dim3(256), 0, stream>>>(gates, pc, hN);

  // heads -> d_out = [logits | values]
  gemm(hN, 256, ppw, 8, p_b, out,           18, NB, 18, 0, 0, nullptr, 0);
  gemm(hN, 256, pvw, 8, v_b, out + NB * 18, 18, NB, 18, 0, 0, nullptr, 0);
}
